// EvoBinarizedLayer_84121229459704
// MI455X (gfx1250) — compile-verified, matrix operand order checked by reading, not run on hardware
//
#include <hip/hip_runtime.h>

// Problem dims (P, B, I, O) = (16, 512, 2048, 2048)
#define P_DIM 16
#define B_DIM 512
#define I_DIM 2048
#define O_DIM 2048

typedef __attribute__((ext_vector_type(8))) int v8i;
typedef __attribute__((ext_vector_type(4))) int v4i;

union F8 {
  v8i v;
  unsigned long long q[4];
};

#define GLOBAL_AS __attribute__((address_space(1)))
#define LDS_AS __attribute__((address_space(3)))

#if defined(__has_builtin)
#if __has_builtin(__builtin_amdgcn_global_load_async_to_lds_b128)
#define USE_ASYNC_LDS 1
#endif
#endif

// Workspace layout (bytes):
//   [0, WDT)            : wdT  = i8 (w0-w1) transposed, [p][o][i]     (64 MiB)
//   [WDT, WDT+XQ)       : xq   = u8 x, [p][b][i]                      (16 MiB)
//   [WDT+XQ, ...)       : bias = f32 sum_i w1, [p][o]                 (128 KiB)
#define WDT_BYTES ((size_t)P_DIM * O_DIM * I_DIM)
#define XQ_BYTES  ((size_t)P_DIM * B_DIM * I_DIM)
#define BIAS_OFF  (WDT_BYTES + XQ_BYTES)

// ---------------------------------------------------------------- bias zero
__global__ void zero_bias_kernel(float* __restrict__ bias) {
  int t = blockIdx.x * blockDim.x + threadIdx.x;
  if (t < P_DIM * O_DIM) bias[t] = 0.0f;
}

// ------------------------------------------------------------- x -> u8 pack
__global__ void quantize_x_kernel(const float* __restrict__ x,
                                  unsigned char* __restrict__ xq) {
  int t = blockIdx.x * blockDim.x + threadIdx.x;  // each handles 16 floats
  const float4* xv = (const float4*)x + (size_t)t * 4;
  unsigned int o[4];
#pragma unroll
  for (int j = 0; j < 4; ++j) {
    float4 f = xv[j];
    o[j] = (unsigned int)(unsigned char)f.x |
           ((unsigned int)(unsigned char)f.y << 8) |
           ((unsigned int)(unsigned char)f.z << 16) |
           ((unsigned int)(unsigned char)f.w << 24);
  }
  ((uint4*)xq)[t] = make_uint4(o[0], o[1], o[2], o[3]);
}

// ---------------------- w -> i8 diff transposed ([p][o][i]) + bias = sum w1
__global__ void prep_w_kernel(const float* __restrict__ w,
                              signed char* __restrict__ wdT,
                              float* __restrict__ bias) {
  __shared__ signed char tile[32][33];
  __shared__ float sb[8][32];
  const int tx = threadIdx.x;  // 0..31
  const int ty = threadIdx.y;  // 0..7
  const int i0 = blockIdx.x * 32;
  const int o0 = blockIdx.y * 32;
  const int p = blockIdx.z;
  const float* w0 = w + (size_t)p * I_DIM * O_DIM;
  const float* w1 = w + ((size_t)P_DIM + p) * I_DIM * O_DIM;

  float bsum = 0.0f;
#pragma unroll
  for (int r = 0; r < 4; ++r) {
    int i = i0 + ty + r * 8;
    size_t idx = (size_t)i * O_DIM + (o0 + tx);
    float a = w0[idx];
    float b = w1[idx];
    tile[ty + r * 8][tx] = (signed char)((int)a - (int)b);
    bsum += b;
  }
  sb[ty][tx] = bsum;
  __syncthreads();
  if (ty == 0) {
    float s = 0.0f;
#pragma unroll
    for (int j = 0; j < 8; ++j) s += sb[j][tx];
    atomicAdd(&bias[p * O_DIM + o0 + tx], s);
  }
#pragma unroll
  for (int r = 0; r < 4; ++r) {
    int o = o0 + ty + r * 8;
    wdT[((size_t)p * O_DIM + o) * I_DIM + (i0 + tx)] = tile[tx][ty + r * 8];
  }
}

// --------------------------------------------------- int8 WMMA GEMM + bias
// out[p, m, o] = bias[p, o] + sum_k xq[p, m, k] * wdT[p, o, k]
#define MT 128
#define NT 128
#define KT 64
#define LDS_STRIDE 80                  // 64B rows padded to 80 (16B aligned)
#define TILE_BYTES (128 * LDS_STRIDE)  // 10240 B per staged tile

__global__ __launch_bounds__(256) void bingemm_wmma_kernel(
    const unsigned char* __restrict__ xq, const signed char* __restrict__ wdT,
    const float* __restrict__ bias, float* __restrict__ out) {
  __shared__ __align__(16) unsigned char smem[4 * TILE_BYTES];  // A0 B0 A1 B1

  const int tid = threadIdx.x;
  const int lane = tid & 31;
  const int wave = tid >> 5;             // 0..7
  const int wm = wave >> 2;              // 0..1  (wave row: 64 M each)
  const int wn = wave & 3;               // 0..3  (wave col: 32 N each)
  const int lr = lane & 15;              // fragment row within 16
  const int hk = ((lane >> 4) & 1) * 8;  // byte offset for lanes 16..31
  const int p = blockIdx.z;
  const int m0 = blockIdx.y * MT;
  const int n0 = blockIdx.x * NT;

  const unsigned char* Ag = xq + (size_t)p * B_DIM * I_DIM;
  const unsigned char* Bg =
      (const unsigned char*)wdT + (size_t)p * O_DIM * I_DIM;

  v8i acc[4][2] = {};

  const int srow = tid >> 2;        // 0..63 (plus +64 for second vector)
  const int sseg = (tid & 3) * 16;  // 16B segment within a 64B row

#if defined(USE_ASYNC_LDS)
  // CDNA5 async copy: global -> LDS without touching VGPR data (ASYNCcnt)
  auto stage = [&](int kc, int buf) {
    unsigned char* la = smem + buf * 2 * TILE_BYTES;
    unsigned char* lb = la + TILE_BYTES;
#pragma unroll
    for (int j = 0; j < 2; ++j) {
      int row = srow + j * 64;
      size_t koff = (size_t)kc * KT + sseg;
      int lofs = row * LDS_STRIDE + sseg;
      __builtin_amdgcn_global_load_async_to_lds_b128(
          (GLOBAL_AS v4i*)(Ag + (size_t)(m0 + row) * I_DIM + koff),
          (LDS_AS v4i*)(la + lofs), 0, 0);
      __builtin_amdgcn_global_load_async_to_lds_b128(
          (GLOBAL_AS v4i*)(Bg + (size_t)(n0 + row) * I_DIM + koff),
          (LDS_AS v4i*)(lb + lofs), 0, 0);
    }
  };
  auto wait_stage = [&]() {
#if __has_builtin(__builtin_amdgcn_s_wait_asynccnt)
    __builtin_amdgcn_s_wait_asynccnt(0);
#else
    asm volatile("s_wait_asynccnt 0" ::: "memory");
#endif
  };
#else
  int4 ra[2], rb[2];
  auto ldregs = [&](int kc) {
#pragma unroll
    for (int j = 0; j < 2; ++j) {
      int row = srow + j * 64;
      size_t koff = (size_t)kc * KT + sseg;
      ra[j] = *(const int4*)(Ag + (size_t)(m0 + row) * I_DIM + koff);
      rb[j] = *(const int4*)(Bg + (size_t)(n0 + row) * I_DIM + koff);
    }
  };
  auto store_lds = [&](int buf) {
    unsigned char* la = smem + buf * 2 * TILE_BYTES;
    unsigned char* lb = la + TILE_BYTES;
#pragma unroll
    for (int j = 0; j < 2; ++j) {
      int off = (srow + j * 64) * LDS_STRIDE + sseg;
      *(int4*)(la + off) = ra[j];
      *(int4*)(lb + off) = rb[j];
    }
  };
#endif

  // 8-bit 16x64 A/B fragment: lanes 0-15 bytes {0-7,16-23,32-39,48-55} of the
  // row, lanes 16-31 the same +8. Four ds_load_b64 per fragment.
  auto load_frag = [&](const unsigned char* base, int fragRow, F8& f) {
    const unsigned long long* pq =
        (const unsigned long long*)(base + fragRow * LDS_STRIDE + hk);
    f.q[0] = pq[0];
    f.q[1] = pq[2];
    f.q[2] = pq[4];
    f.q[3] = pq[6];
  };
  auto compute = [&](int buf) {
    const unsigned char* la = smem + buf * 2 * TILE_BYTES;
    const unsigned char* lb = la + TILE_BYTES;
    F8 af[4];
    F8 bf[2];
#pragma unroll
    for (int fm = 0; fm < 4; ++fm)
      load_frag(la, wm * 64 + fm * 16 + lr, af[fm]);
#pragma unroll
    for (int fn = 0; fn < 2; ++fn)
      load_frag(lb, wn * 32 + fn * 16 + lr, bf[fn]);
#pragma unroll
    for (int fm = 0; fm < 4; ++fm)
#pragma unroll
      for (int fn = 0; fn < 2; ++fn)
        acc[fm][fn] = __builtin_amdgcn_wmma_i32_16x16x64_iu8(
            /*sgn_a=*/false, af[fm].v, /*sgn_b=*/true, bf[fn].v, acc[fm][fn],
            /*reuse_a=*/false, /*reuse_b=*/false);
  };

  const int NK = I_DIM / KT;  // 32 K-chunks (power of two)

#if defined(USE_ASYNC_LDS)
  stage(0, 0);
  wait_stage();
  __syncthreads();
  for (int kc = 0; kc < NK; kc += 2) {
    stage(kc + 1, 1);  // overlap with compute on buf0
    compute(0);
    wait_stage();
    __syncthreads();
    stage((kc + 2) & (NK - 1), 0);  // last one wraps (harmless dummy)
    compute(1);
    wait_stage();
    __syncthreads();
  }
#else
  ldregs(0);
  store_lds(0);
  __syncthreads();
  for (int kc = 0; kc < NK; kc += 2) {
    ldregs(kc + 1);  // issue global loads early
    compute(0);
    store_lds(1);
    __syncthreads();
    ldregs((kc + 2) & (NK - 1));  // last one wraps (harmless dummy)
    compute(1);
    store_lds(0);
    __syncthreads();
  }
#endif

  // Epilogue: D layout (32-bit 16x16): lane<16 -> M=g, N=lane; lane>=16 -> M=g+8
#pragma unroll
  for (int fn = 0; fn < 2; ++fn) {
    int ocol = n0 + wn * 32 + fn * 16 + lr;
    float bv = bias[p * O_DIM + ocol];
#pragma unroll
    for (int fm = 0; fm < 4; ++fm) {
      int mrow = m0 + wm * 64 + fm * 16 + ((lane >> 4) & 1) * 8;
      float* op = out + ((size_t)p * B_DIM + mrow) * O_DIM + ocol;
#pragma unroll
      for (int g = 0; g < 8; ++g)
        op[(size_t)g * O_DIM] = bv + (float)acc[fm][fn][g];
    }
  }
}

// --------------------------------------------------------------------------
extern "C" void kernel_launch(void* const* d_in, const int* in_sizes, int n_in,
                              void* d_out, int out_size, void* d_ws,
                              size_t ws_size, hipStream_t stream) {
  (void)in_sizes;
  (void)n_in;
  (void)out_size;
  (void)ws_size;
  const float* x = (const float*)d_in[0];
  const float* w = (const float*)d_in[1];
  float* out = (float*)d_out;

  signed char* wdT = (signed char*)d_ws;
  unsigned char* xq = (unsigned char*)d_ws + WDT_BYTES;
  float* bias = (float*)((char*)d_ws + BIAS_OFF);

  zero_bias_kernel<<<(P_DIM * O_DIM + 255) / 256, 256, 0, stream>>>(bias);
  prep_w_kernel<<<dim3(I_DIM / 32, O_DIM / 32, P_DIM), dim3(32, 8), 0,
                  stream>>>(w, wdT, bias);
  quantize_x_kernel<<<(int)((XQ_BYTES / 16 + 255) / 256), 256, 0, stream>>>(x,
                                                                            xq);
  bingemm_wmma_kernel<<<dim3(O_DIM / NT, B_DIM / MT, P_DIM), 256, 0, stream>>>(
      xq, wdT, bias, out);
}
